// DGPN2Cls_89945205112960
// MI455X (gfx1250) — matrix operand order repaired and checked by reading, not compiled, not run on hardware
//
#include <hip/hip_runtime.h>

#define B_   32
#define N_   4096
#define NP_  256
#define NS_  64
#define KNN_ 20
#define R2_  0.04f

typedef __attribute__((ext_vector_type(16))) __bf16 v16bf;
typedef __attribute__((ext_vector_type(8)))  __bf16 v8bf;
typedef __attribute__((ext_vector_type(8)))  float  v8f;

// ---------------------------------------------------------------- helpers
__device__ __forceinline__ __bf16 f2bf(float f){
  unsigned u = __float_as_uint(f);
  u += 0x7FFFu + ((u >> 16) & 1u);
  unsigned short h = (unsigned short)(u >> 16);
  return __builtin_bit_cast(__bf16, h);
}

// A fragment (16x32) from bf16 row-major; pitch and k0 multiples of 32 -> 16B aligned.
// lane<16: row m0+lane, K = k0+{0..7,16..23}; lane>=16: row m0+lane-16, K = k0+{8..15,24..31}
__device__ __forceinline__ v16bf load_a_bf(const __bf16* A, long lda, int m0, int k0, int lane){
  const __bf16* Ar = A + (long)(m0 + (lane & 15)) * lda + k0 + ((lane & 16) ? 8 : 0);
  v8bf lo = *(const v8bf*)(Ar);
  v8bf hi = *(const v8bf*)(Ar + 16);
  return __builtin_shufflevector(lo, hi, 0,1,2,3,4,5,6,7,8,9,10,11,12,13,14,15);
}

// B fragment (32x16), B = W^T with W (N x K) bf16 row-major.
// lane<16: col n0+lane, K = k0+{0..7,16..23}; lane>=16: col n0+lane-16, K = k0+{8..15,24..31}
__device__ __forceinline__ v16bf load_b_bf(const __bf16* W, long ldw, int n0, int k0, int lane){
  const __bf16* Wr = W + (long)(n0 + (lane & 15)) * ldw + k0 + ((lane & 16) ? 8 : 0);
  v8bf b0 = *(const v8bf*)(Wr);
  v8bf b1 = *(const v8bf*)(Wr + 16);
  return __builtin_shufflevector(b0, b1, 0,1,2,3,4,5,6,7,8,9,10,11,12,13,14,15);
}

__device__ __forceinline__ v8f wmma_bf(v16bf a, v16bf b, v8f c){
  return __builtin_amdgcn_wmma_f32_16x16x32_bf16(false, a, false, b, (short)0, c, false, false);
}

// epilogue for one 16x16 sub-tile
__device__ __forceinline__ void store_tile(v8f acc, float* Cf, __bf16* Ch, long ldc,
                                           int mm, int nn, int lane,
                                           int relu, int mode, int gdiv){
  int c  = nn + (lane & 15);
  int rb = mm + ((lane & 16) ? 8 : 0);
#pragma unroll
  for (int r = 0; r < 8; ++r){
    float v = acc[r];
    if (relu) v = v > 0.f ? v : 0.f;
    if (mode == 0){
      if (Cf) Cf[(long)(rb + r) * ldc + c] = v;
      if (Ch) Ch[(long)(rb + r) * ldc + c] = f2bf(v);
    } else {
      int g = (rb + r) / gdiv;
      atomicMax((int*)(Cf + (long)g * ldc + c), __float_as_int(v));
    }
  }
}

// ---------------------------------------------------------------- generic WMMA GEMM
// [relu](A @ W^T). K multiple of 32 (operands zero-padded by producers);
// M, N multiples of 32. One wave computes a 32x32 macro-tile (2x2 register
// blocking: 4 WMMAs per 2 A-frag + 2 B-frag loads -> 2x arithmetic intensity).
// mode 0: store fp32 to Cf (if non-null) and/or bf16 to Ch (if non-null).
// mode 1: atomicMax of relu output into pooled Cf[row/gdiv][n] (pre-zeroed; relu>=0).
__global__ void __launch_bounds__(256)
k_gemm(const __bf16* A, long lda, long strideA,
       const __bf16* W, long ldw, long strideW,
       float* Cf, __bf16* Ch, long ldc, long strideC,
       int M, int N, int K, int relu, int mode, int gdiv){
  int lane = threadIdx.x & 31;
  int wave = threadIdx.x >> 5;
  int ntn = N >> 5, ntm = M >> 5;
  long tile  = (long)blockIdx.x * 8 + wave;
  long total = (long)ntm * ntn;
  if (tile >= total) return;
  int mt = (int)(tile / ntn), nt = (int)(tile % ntn);
  int b = blockIdx.y;
  A += (long)b * strideA;  W += (long)b * strideW;
  if (Cf) Cf += (long)b * strideC;
  if (Ch) Ch += (long)b * strideC;
  int m0 = mt << 5, n0 = nt << 5;
  v8f acc00 = {0.f,0.f,0.f,0.f,0.f,0.f,0.f,0.f};
  v8f acc01 = acc00, acc10 = acc00, acc11 = acc00;
  for (int k0 = 0; k0 < K; k0 += 32){
    v16bf a0 = load_a_bf(A, lda, m0,      k0, lane);
    v16bf a1 = load_a_bf(A, lda, m0 + 16, k0, lane);
    v16bf b0 = load_b_bf(W, ldw, n0,      k0, lane);
    v16bf b1 = load_b_bf(W, ldw, n0 + 16, k0, lane);
    acc00 = wmma_bf(a0, b0, acc00);
    acc01 = wmma_bf(a0, b1, acc01);
    acc11 = wmma_bf(a1, b1, acc11);
    acc10 = wmma_bf(a1, b0, acc10);
  }
  store_tile(acc00, Cf, Ch, ldc, m0,      n0,      lane, relu, mode, gdiv);
  store_tile(acc01, Cf, Ch, ldc, m0,      n0 + 16, lane, relu, mode, gdiv);
  store_tile(acc10, Cf, Ch, ldc, m0 + 16, n0,      lane, relu, mode, gdiv);
  store_tile(acc11, Cf, Ch, ldc, m0 + 16, n0 + 16, lane, relu, mode, gdiv);
}

// ---------------------------------------------------------------- stage kernels
__global__ void k_transpose(const float* __restrict__ pts, float* __restrict__ xyz){
  int i = blockIdx.x * 256 + threadIdx.x;
  if (i >= B_ * N_) return;
  int b = i / N_, n = i % N_;
  const float* s = pts + (long)b * 3 * N_;
  float* d = xyz + (long)i * 3;
  d[0] = s[n]; d[1] = s[N_ + n]; d[2] = s[2L * N_ + n];
}

__global__ void __launch_bounds__(256)
k_fps(const float* __restrict__ xyz, int* __restrict__ fps_idx, float* __restrict__ new_xyz){
  __shared__ float dist[N_];
  __shared__ float rv[256];
  __shared__ int   ri[256];
  __shared__ int   ssel[NP_];
  __shared__ int   s_last;
  int b = blockIdx.x, t = threadIdx.x;
  const float* X = xyz + (long)b * N_ * 3;
  for (int j = t; j < N_; j += 256) dist[j] = 3.4e38f;
  int last = 0;
  __syncthreads();
  for (int it = 0; it < NP_; ++it){
    if (t == 0){ fps_idx[b * NP_ + it] = last; ssel[it] = last; }
    float px = X[last * 3], py = X[last * 3 + 1], pz = X[last * 3 + 2];
    float bv = -1.f; int bi = 0;
    for (int j = t; j < N_; j += 256){
      float dx = X[j*3] - px, dy = X[j*3+1] - py, dz = X[j*3+2] - pz;
      float d = dx*dx + dy*dy + dz*dz;
      float dm = dist[j];
      d = d < dm ? d : dm;
      dist[j] = d;
      if (d > bv){ bv = d; bi = j; }   // ascending j: strict > keeps lowest idx on ties
    }
    rv[t] = bv; ri[t] = bi;
    __syncthreads();
    for (int s = 128; s > 0; s >>= 1){
      if (t < s){
        float ov = rv[t+s]; int oi = ri[t+s];
        if (ov > rv[t] || (ov == rv[t] && oi < ri[t])){ rv[t] = ov; ri[t] = oi; }
      }
      __syncthreads();
    }
    if (t == 0) s_last = ri[0];
    __syncthreads();
    last = s_last;
    __syncthreads();
  }
  if (t < NP_){
    int id = ssel[t];
    float* d = new_xyz + ((long)b * NP_ + t) * 3;
    d[0] = X[id*3]; d[1] = X[id*3+1]; d[2] = X[id*3+2];
  }
}

__global__ void __launch_bounds__(256)
k_ball(const float* __restrict__ xyz, const float* __restrict__ new_xyz,
       float* __restrict__ grouped){
  __shared__ int sidx[8][NS_];
  int lane = threadIdx.x & 31, wv = threadIdx.x >> 5;
  long gw = (long)blockIdx.x * 8 + wv;
  if (gw >= (long)B_ * NP_) return;
  int b = (int)(gw / NP_);
  const float* X = xyz + (long)b * N_ * 3;
  float cx = new_xyz[gw*3], cy = new_xyz[gw*3+1], cz = new_xyz[gw*3+2];
  int cnt = 0;
  for (int base = 0; base < N_ && cnt < NS_; base += 32){
    int j = base + lane;
    float dx = X[j*3] - cx, dy = X[j*3+1] - cy, dz = X[j*3+2] - cz;
    bool in = (dx*dx + dy*dy + dz*dz) < R2_;
    unsigned mask = (unsigned)__ballot(in);
    if (in){
      int pos = cnt + __popc(mask & ((1u << lane) - 1u));
      if (pos < NS_) sidx[wv][pos] = j;
    }
    cnt += __popc(mask);
  }
  if (cnt > NS_) cnt = NS_;
  for (int s = lane; s < NS_; s += 32){
    int j = (s < cnt) ? sidx[wv][s] : sidx[wv][0];
    long o = (gw * NS_ + s) * 3;
    grouped[o]   = X[j*3]   - cx;
    grouped[o+1] = X[j*3+1] - cy;
    grouped[o+2] = X[j*3+2] - cz;
  }
}

// fused SA MLP per group: 64x3 -> 64 -> 64 -> 128, max over 64 samples
__global__ void __launch_bounds__(128)
k_sa_fused(const float* __restrict__ grouped, const float* __restrict__ sa_w0,
           const __bf16* __restrict__ w1h, const __bf16* __restrict__ w2h,
           float* __restrict__ feat){
  __shared__ float  sX[NS_ * 3];
  __shared__ float  sW0[64 * 3];
  __shared__ __align__(16) __bf16 Ha[64 * 64];
  __shared__ __align__(16) __bf16 Hb[64 * 64];
  int g = blockIdx.x, t = threadIdx.x;
  int lane = t & 31, wv = t >> 5;
  for (int i = t; i < NS_ * 3; i += 128) sX[i]  = grouped[(long)g * (NS_*3) + i];
  for (int i = t; i < 64 * 3;  i += 128) sW0[i] = sa_w0[i];
  __syncthreads();
  for (int i = t; i < 64 * 64; i += 128){
    int s = i >> 6, o = i & 63;
    float v = sX[s*3]*sW0[o*3] + sX[s*3+1]*sW0[o*3+1] + sX[s*3+2]*sW0[o*3+2];
    Ha[s*64 + o] = f2bf(v > 0.f ? v : 0.f);
  }
  __syncthreads();
  { // layer1: H2 = relu(H1 @ W1^T), 64x64, K=64
    int m0 = wv << 4;
    for (int nt = 0; nt < 4; ++nt){
      v8f acc = {0.f,0.f,0.f,0.f,0.f,0.f,0.f,0.f};
      for (int kt = 0; kt < 2; ++kt){
        v16bf a = load_a_bf(Ha, 64, m0, kt*32, lane);
        v16bf bfr = load_b_bf(w1h, 64, nt*16, kt*32, lane);
        acc = wmma_bf(a, bfr, acc);
      }
      int c  = (nt << 4) + (lane & 15);
      int rb = m0 + ((lane & 16) ? 8 : 0);
#pragma unroll
      for (int r = 0; r < 8; ++r){
        float v = acc[r]; v = v > 0.f ? v : 0.f;
        Hb[(rb + r)*64 + c] = f2bf(v);
      }
    }
  }
  __syncthreads();
  { // layer2: 64x128, K=64, fused column max over 64 samples
    for (int nt2 = 0; nt2 < 2; ++nt2){
      int n0 = (wv << 5) + (nt2 << 4);
      v8f vmax = {0.f,0.f,0.f,0.f,0.f,0.f,0.f,0.f};
      for (int mt = 0; mt < 4; ++mt){
        v8f acc = {0.f,0.f,0.f,0.f,0.f,0.f,0.f,0.f};
        for (int kt = 0; kt < 2; ++kt){
          v16bf a = load_a_bf(Hb, 64, mt*16, kt*32, lane);
          v16bf bfr = load_b_bf(w2h, 64, n0, kt*32, lane);
          acc = wmma_bf(a, bfr, acc);
        }
#pragma unroll
        for (int r = 0; r < 8; ++r){
          float v = acc[r]; v = v > 0.f ? v : 0.f;
          vmax[r] = vmax[r] > v ? vmax[r] : v;
        }
      }
      float m = 0.f;
#pragma unroll
      for (int r = 0; r < 8; ++r) m = m > vmax[r] ? m : vmax[r];
      float o = __shfl_xor(m, 16, 32);
      m = m > o ? m : o;
      if (lane < 16) feat[(long)g * 128 + n0 + lane] = m;
    }
  }
}

// xh(bf16, pitch 160) = concat(feat[128], (useT? T@new_xyz : new_xyz)[3], zeros[29]); + norms
__global__ void __launch_bounds__(256)
k_xform_concat(const float* __restrict__ feat, const float* __restrict__ new_xyz,
               const float* __restrict__ tmat, int useT,
               __bf16* __restrict__ xh, float* __restrict__ nrm){
  __shared__ float red[256];
  int g = blockIdx.x, t = threadIdx.x;
  int b = g / NP_;
  float v = 0.f, ss = 0.f;
  if (t < 128) v = feat[(long)g * 128 + t];
  else if (t < 131){
    int i3 = t - 128;
    const float* nx = new_xyz + (long)g * 3;
    if (useT){
      const float* T = tmat + b * 9;
      v = T[i3*3]*nx[0] + T[i3*3+1]*nx[1] + T[i3*3+2]*nx[2];
    } else v = nx[i3];
  }
  if (t < 160) xh[(long)g * 160 + t] = f2bf(v);
  if (t < 131) ss = v * v;
  red[t] = ss;
  __syncthreads();
  for (int s = 128; s > 0; s >>= 1){ if (t < s) red[t] += red[t+s]; __syncthreads(); }
  if (t == 0) nrm[g] = red[0];
}

// top-KNN_ smallest d = nrm[i]+nrm[j]-2G[i][j], ties -> lowest index (lexicographic scan)
__global__ void k_topk(const float* __restrict__ gram, const float* __restrict__ nrm,
                       int* __restrict__ knn){
  int i = blockIdx.x * 256 + threadIdx.x;
  if (i >= B_ * NP_) return;
  int b = i / NP_;
  const float* G  = gram + (long)i * NP_;
  const float* nr = nrm + (long)b * NP_;
  float ni = nrm[i];
  float prevd = -3.4e38f; int prevj = -1;
  for (int kk = 0; kk < KNN_; ++kk){
    float bd = 3.4e38f; int bj = 0;
    for (int j = 0; j < NP_; ++j){
      float d = ni + nr[j] - 2.f * G[j];
      bool gt = (d > prevd) || (d == prevd && j > prevj);
      if (gt && (d < bd || (d == bd && j < bj))){ bd = d; bj = j; }
    }
    knn[(long)i * KNN_ + kk] = bj;
    prevd = bd; prevj = bj;
  }
}

// split W (Cout x 2Cin) -> Wd = Wc - Wn, Wn, bf16 with zero-padded pitch KP
__global__ void k_wsplit(const float* __restrict__ W, int Cout, int Cin, int KP,
                         __bf16* __restrict__ Wd, __bf16* __restrict__ Wn){
  int i = blockIdx.x * 256 + threadIdx.x;
  if (i >= Cout * KP) return;
  int o = i / KP, c = i % KP;
  float wc = 0.f, wn = 0.f;
  if (c < Cin){
    wc = W[(long)o * 2 * Cin + c];
    wn = W[(long)o * 2 * Cin + Cin + c];
  }
  Wd[i] = f2bf(wc - wn);
  Wn[i] = f2bf(wn);
}

// generic fp32 -> bf16 convert
__global__ void k_cvtbf(const float* __restrict__ in, __bf16* __restrict__ out, long n){
  long i = (long)blockIdx.x * 256 + threadIdx.x;
  if (i < n) out[i] = f2bf(in[i]);
}

// tnet: h1[g,jn,o] = relu(A1[g,o] + A2[nbr,o]) in bf16
__global__ void k_tnet_h1(const float* __restrict__ A1, const float* __restrict__ A2,
                          const int* __restrict__ knn, __bf16* __restrict__ h1, long total){
  long i = (long)blockIdx.x * 256 + threadIdx.x;
  if (i >= total) return;
  int  o  = (int)(i & 63);
  long t  = i >> 6;
  int  jn = (int)(t % KNN_);
  long g  = t / KNN_;
  int  b  = (int)(g / NP_);
  long nb = (long)b * NP_ + knn[g * KNN_ + jn];
  float v = A1[g * 64 + o] + A2[nb * 64 + o];
  h1[i] = f2bf(v > 0.f ? v : 0.f);
}

// dg block: out[g,o] = relu(A1[g,o] + max_j A2[nbr_j,o]) in bf16; also next-block norms
__global__ void __launch_bounds__(256)
k_dgcombine(const float* __restrict__ A1, const float* __restrict__ A2,
            const int* __restrict__ knn, __bf16* __restrict__ outc, long ldo,
            float* __restrict__ nrm_out, int Cout){
  __shared__ float red[256];
  __shared__ long  nb[KNN_];
  int g = blockIdx.x, t = threadIdx.x;
  int b = g / NP_;
  if (t < KNN_) nb[t] = ((long)b * NP_ + knn[(long)g * KNN_ + t]) * (long)Cout;
  __syncthreads();
  float ss = 0.f;
  for (int o = t; o < Cout; o += 256){
    float m = -3.4e38f;
    for (int j = 0; j < KNN_; ++j){
      float v = A2[nb[j] + o];
      m = v > m ? v : m;
    }
    float v = A1[(long)g * Cout + o] + m;
    v = v > 0.f ? v : 0.f;
    outc[(long)g * ldo + o] = f2bf(v);
    ss += v * v;
  }
  red[t] = ss;
  __syncthreads();
  for (int s = 128; s > 0; s >>= 1){ if (t < s) red[t] += red[t+s]; __syncthreads(); }
  if (t == 0 && nrm_out) nrm_out[g] = red[0];
}

__global__ void k_zero(float* __restrict__ p, long n){
  long i = (long)blockIdx.x * 256 + threadIdx.x;
  if (i < n) p[i] = 0.f;
}

__global__ void k_tnet_lin(const float* __restrict__ g, const float* __restrict__ lw,
                           const float* __restrict__ lb, float* __restrict__ tmat){
  int i = blockIdx.x * 256 + threadIdx.x;
  if (i >= B_ * 9) return;
  int b = i / 9, o = i % 9;
  float s = lb[o];
  for (int c = 0; c < 256; ++c) s += g[b * 256 + c] * lw[o * 256 + c];
  int r = o / 3, cc = o % 3;
  tmat[i] = s + (r == cc ? 1.f : 0.f);
}

__global__ void k_cls(const float* __restrict__ g, const float* __restrict__ W,
                      const float* __restrict__ bias, float* __restrict__ out){
  int i = blockIdx.x * 256 + threadIdx.x;
  if (i >= B_ * 40) return;
  int b = i / 40, o = i % 40;
  float s = bias[o];
  for (int c = 0; c < 256; ++c) s += g[b * 256 + c] * W[o * 256 + c];
  out[i] = s;
}

// ---------------------------------------------------------------- workspace layout
// All offsets in float (4-byte) units, every region a multiple of 32 floats -> 16B+ aligned.
// bf16 regions store 2 elements per float slot.
static const long OFF_XYZ    = 0;
static const long OFF_NEWXYZ = OFF_XYZ    + (long)B_ * N_ * 3;
static const long OFF_FPS    = OFF_NEWXYZ + (long)B_ * NP_ * 3;
static const long OFF_GRP    = OFF_FPS    + (long)B_ * NP_;
static const long OFF_FEAT   = OFF_GRP    + (long)B_ * NP_ * NS_ * 3;
static const long OFF_NRM    = OFF_FEAT   + (long)B_ * NP_ * 128;
static const long OFF_GRAM   = OFF_NRM    + (long)B_ * NP_;
static const long OFF_KNN    = OFF_GRAM   + (long)B_ * NP_ * NP_;
static const long OFF_A1     = OFF_KNN    + (long)B_ * NP_ * KNN_;
static const long OFF_A2     = OFF_A1     + (long)B_ * NP_ * 512;
static const long OFF_TFEAT  = OFF_A2     + (long)B_ * NP_ * 512;
static const long OFF_TGLOB  = OFF_TFEAT  + (long)B_ * NP_ * 128;
static const long OFF_TG2    = OFF_TGLOB  + (long)B_ * 1024;
static const long OFF_TMAT   = OFF_TG2    + (long)B_ * 256;
static const long OFF_DGLOB  = OFF_TMAT   + (long)B_ * 9;        // 288, mult of 32
static const long OFF_DG2    = OFF_DGLOB  + (long)B_ * 128;
// bf16 regions (sizes below are in float units = bf16_count/2)
static const long OFF_XH     = OFF_DG2    + (long)B_ * 256;
static const long OFF_CATH   = OFF_XH     + (long)B_ * NP_ * 160 / 2;
static const long OFF_H1H    = OFF_CATH   + (long)B_ * NP_ * 896 / 2;
static const long OFF_WDH    = OFF_H1H    + (long)B_ * NP_ * KNN_ * 64 / 2;
static const long OFF_WNH    = OFF_WDH    + 512L * 256 / 2;
static const long OFF_TFEATH = OFF_WNH    + 512L * 256 / 2;
static const long OFF_TGLOBH = OFF_TFEATH + (long)B_ * NP_ * 128 / 2;
static const long OFF_TG1H   = OFF_TGLOBH + (long)B_ * 1024 / 2;
static const long OFF_DGLOBH = OFF_TG1H   + (long)B_ * 512 / 2;
static const long OFF_DG1H   = OFF_DGLOBH + (long)B_ * 128 / 2;
static const long OFF_WSA1   = OFF_DG1H   + (long)B_ * 512 / 2;
static const long OFF_WSA2   = OFF_WSA1   + 64L * 64 / 2;
static const long OFF_WTEC1  = OFF_WSA2   + 128L * 64 / 2;
static const long OFF_WTLOC  = OFF_WTEC1  + 128L * 64 / 2;
static const long OFF_WTG0   = OFF_WTLOC  + 1024L * 128 / 2;
static const long OFF_WTG1   = OFF_WTG0   + 512L * 1024 / 2;
static const long OFF_WDGL   = OFF_WTG1   + 256L * 512 / 2;
static const long OFF_WDGG0  = OFF_WDGL   + 128L * 896 / 2;
static const long OFF_WDGG1  = OFF_WDGG0  + 512L * 128 / 2;

static inline void gemm(hipStream_t s, const __bf16* A, long lda, long sA,
                        const __bf16* W, long ldw, long sW,
                        float* Cf, __bf16* Ch, long ldc, long sC,
                        int M, int N, int K, int batches,
                        int relu, int mode, int gdiv){
  long tiles = (long)(M / 32) * (N / 32);
  dim3 grid((unsigned)((tiles + 7) / 8), (unsigned)batches, 1);
  k_gemm<<<grid, 256, 0, s>>>(A, lda, sA, W, ldw, sW, Cf, Ch, ldc, sC,
                              M, N, K, relu, mode, gdiv);
}

static inline void cvtbf(hipStream_t s, const float* in, __bf16* out, long n){
  k_cvtbf<<<(unsigned)((n + 255) / 256), 256, 0, s>>>(in, out, n);
}

extern "C" void kernel_launch(void* const* d_in, const int* in_sizes, int n_in,
                              void* d_out, int out_size, void* d_ws, size_t ws_size,
                              hipStream_t stream){
  (void)in_sizes; (void)n_in; (void)out_size; (void)ws_size;
  const float* points       = (const float*)d_in[0];
  const float* sa_w0        = (const float*)d_in[1];
  const float* sa_w1        = (const float*)d_in[2];
  const float* sa_w2        = (const float*)d_in[3];
  const float* tnet_ec_w0   = (const float*)d_in[4];
  const float* tnet_ec_w1   = (const float*)d_in[5];
  const float* tnet_local_w = (const float*)d_in[6];
  const float* tnet_g_w0    = (const float*)d_in[7];
  const float* tnet_g_w1    = (const float*)d_in[8];
  const float* tnet_lin_w   = (const float*)d_in[9];
  const float* tnet_lin_b   = (const float*)d_in[10];
  const float* dg_ec_w0     = (const float*)d_in[11];
  const float* dg_ec_w1     = (const float*)d_in[12];
  const float* dg_ec_w2     = (const float*)d_in[13];
  const float* dg_local_w   = (const float*)d_in[14];
  const float* dg_g_w0      = (const float*)d_in[15];
  const float* dg_g_w1      = (const float*)d_in[16];
  const float* cls_w        = (const float*)d_in[17];
  const float* cls_b        = (const float*)d_in[18];

  float* ws      = (float*)d_ws;
  float*  xyz     = ws + OFF_XYZ;
  float*  new_xyz = ws + OFF_NEWXYZ;
  int*    fpsidx  = (int*)(ws + OFF_FPS);
  float*  grouped = ws + OFF_GRP;
  float*  feat    = ws + OFF_FEAT;
  float*  nrm     = ws + OFF_NRM;
  float*  gram    = ws + OFF_GRAM;
  int*    knn     = (int*)(ws + OFF_KNN);
  float*  A1      = ws + OFF_A1;
  float*  A2      = ws + OFF_A2;
  float*  tfeat   = ws + OFF_TFEAT;
  float*  tglob   = ws + OFF_TGLOB;
  float*  tg2     = ws + OFF_TG2;
  float*  tmat    = ws + OFF_TMAT;
  float*  dglob   = ws + OFF_DGLOB;
  float*  dg2     = ws + OFF_DG2;
  __bf16* xh      = (__bf16*)(ws + OFF_XH);
  __bf16* cath    = (__bf16*)(ws + OFF_CATH);
  __bf16* h1h     = (__bf16*)(ws + OFF_H1H);
  __bf16* Wdh     = (__bf16*)(ws + OFF_WDH);
  __bf16* Wnh     = (__bf16*)(ws + OFF_WNH);
  __bf16* tfeath  = (__bf16*)(ws + OFF_TFEATH);
  __bf16* tglobh  = (__bf16*)(ws + OFF_TGLOBH);
  __bf16* tg1h    = (__bf16*)(ws + OFF_TG1H);
  __bf16* dglobh  = (__bf16*)(ws + OFF_DGLOBH);
  __bf16* dg1h    = (__bf16*)(ws + OFF_DG1H);
  __bf16* wsa1h   = (__bf16*)(ws + OFF_WSA1);
  __bf16* wsa2h   = (__bf16*)(ws + OFF_WSA2);
  __bf16* wtec1h  = (__bf16*)(ws + OFF_WTEC1);
  __bf16* wtloch  = (__bf16*)(ws + OFF_WTLOC);
  __bf16* wtg0h   = (__bf16*)(ws + OFF_WTG0);
  __bf16* wtg1h   = (__bf16*)(ws + OFF_WTG1);
  __bf16* wdglh   = (__bf16*)(ws + OFF_WDGL);
  __bf16* wdgg0h  = (__bf16*)(ws + OFF_WDGG0);
  __bf16* wdgg1h  = (__bf16*)(ws + OFF_WDGG1);

  // 0) one-time bf16 weight copies (K already multiple of 32 for all of these)
  cvtbf(stream, sa_w1,        wsa1h,  64L*64);
  cvtbf(stream, sa_w2,        wsa2h,  128L*64);
  cvtbf(stream, tnet_ec_w1,   wtec1h, 128L*64);
  cvtbf(stream, tnet_local_w, wtloch, 1024L*128);
  cvtbf(stream, tnet_g_w0,    wtg0h,  512L*1024);
  cvtbf(stream, tnet_g_w1,    wtg1h,  256L*512);
  cvtbf(stream, dg_local_w,   wdglh,  128L*896);
  cvtbf(stream, dg_g_w0,      wdgg0h, 512L*128);
  cvtbf(stream, dg_g_w1,      wdgg1h, 256L*512);

  // 1) geometry: transpose, FPS, ball-query grouping
  k_transpose<<<(B_*N_ + 255)/256, 256, 0, stream>>>(points, xyz);
  k_fps<<<B_, 256, 0, stream>>>(xyz, fpsidx, new_xyz);
  k_ball<<<(B_*NP_ + 7)/8, 256, 0, stream>>>(xyz, new_xyz, grouped);

  // 2) fused SA MLP (3->64->64->128, max over 64)
  k_sa_fused<<<B_*NP_, 128, 0, stream>>>(grouped, sa_w0, wsa1h, wsa2h, feat);

  // 3) T-Net: x = [feat | new_xyz] (pitch 160, zero pad), gram, knn
  k_xform_concat<<<B_*NP_, 256, 0, stream>>>(feat, new_xyz, tmat, 0, xh, nrm);
  gemm(stream, xh, 160, (long)NP_*160, xh, 160, (long)NP_*160,
       gram, nullptr, NP_, (long)NP_*NP_, NP_, NP_, 160, B_, 0, 0, 0);
  k_topk<<<(B_*NP_ + 255)/256, 256, 0, stream>>>(gram, nrm, knn);
  k_wsplit<<<(64*160 + 255)/256, 256, 0, stream>>>(tnet_ec_w0, 64, 131, 160, Wdh, Wnh);
  gemm(stream, xh, 160, 0, Wdh, 160, 0, A1, nullptr, 64, 0, B_*NP_, 64, 160, 1, 0, 0, 0);
  gemm(stream, xh, 160, 0, Wnh, 160, 0, A2, nullptr, 64, 0, B_*NP_, 64, 160, 1, 0, 0, 0);
  {
    long tot = (long)B_*NP_*KNN_*64;
    k_tnet_h1<<<(unsigned)((tot + 255)/256), 256, 0, stream>>>(A1, A2, knn, h1h, tot);
  }
  k_zero<<<(unsigned)(((long)B_*NP_*128 + 255)/256), 256, 0, stream>>>(tfeat, (long)B_*NP_*128);
  gemm(stream, h1h, 64, 0, wtec1h, 64, 0, tfeat, nullptr, 128, 0,
       B_*NP_*KNN_, 128, 64, 1, 1, 1, KNN_);               // relu + max over 20 nbrs
  cvtbf(stream, tfeat, tfeath, (long)B_*NP_*128);
  k_zero<<<(unsigned)(((long)B_*1024 + 255)/256), 256, 0, stream>>>(tglob, (long)B_*1024);
  gemm(stream, tfeath, 128, 0, wtloch, 128, 0, tglob, nullptr, 1024, 0,
       B_*NP_, 1024, 128, 1, 1, 1, NP_);                   // relu + max over 256 pts
  cvtbf(stream, tglob, tglobh, (long)B_*1024);
  gemm(stream, tglobh, 1024, 0, wtg0h, 1024, 0, nullptr, tg1h, 512, 0, B_, 512, 1024, 1, 1, 0, 0);
  gemm(stream, tg1h, 512, 0, wtg1h, 512, 0, tg2, nullptr, 256, 0, B_, 256, 512, 1, 1, 0, 0);
  k_tnet_lin<<<(B_*9 + 255)/256, 256, 0, stream>>>(tg2, tnet_lin_w, tnet_lin_b, tmat);

  // 4) transformed input for DGCNN blocks
  k_xform_concat<<<B_*NP_, 256, 0, stream>>>(feat, new_xyz, tmat, 1, xh, nrm);

  // 5) three edge-conv blocks (collapsed via relu/max distributivity)
  const float*  dgw[3]  = {dg_ec_w0, dg_ec_w1, dg_ec_w2};
  int cin[3]            = {131, 128, 256};
  int kp[3]             = {160, 128, 256};
  int cout[3]           = {128, 256, 512};
  const __bf16* xsrc[3] = {xh, cath + 0, cath + 128};
  long xlda[3]          = {160, 896, 896};
  long colofs[3]        = {0, 128, 384};
  for (int blk = 0; blk < 3; ++blk){
    gemm(stream, xsrc[blk], xlda[blk], (long)NP_*xlda[blk],
         xsrc[blk], xlda[blk], (long)NP_*xlda[blk],
         gram, nullptr, NP_, (long)NP_*NP_, NP_, NP_, kp[blk], B_, 0, 0, 0);
    k_topk<<<(B_*NP_ + 255)/256, 256, 0, stream>>>(gram, nrm, knn);
    k_wsplit<<<(cout[blk]*kp[blk] + 255)/256, 256, 0, stream>>>(dgw[blk], cout[blk], cin[blk],
                                                                kp[blk], Wdh, Wnh);
    gemm(stream, xsrc[blk], xlda[blk], 0, Wdh, kp[blk], 0, A1, nullptr, cout[blk], 0,
         B_*NP_, cout[blk], kp[blk], 1, 0, 0, 0);
    gemm(stream, xsrc[blk], xlda[blk], 0, Wnh, kp[blk], 0, A2, nullptr, cout[blk], 0,
         B_*NP_, cout[blk], kp[blk], 1, 0, 0, 0);
    k_dgcombine<<<B_*NP_, 256, 0, stream>>>(A1, A2, knn, cath + colofs[blk], 896,
                                            (blk < 2) ? nrm : (float*)nullptr, cout[blk]);
  }

  // 6) head: dg_local (relu + max over points), two MLPs, classifier
  k_zero<<<(unsigned)(((long)B_*128 + 255)/256), 256, 0, stream>>>(dglob, (long)B_*128);
  gemm(stream, cath, 896, 0, wdglh, 896, 0, dglob, nullptr, 128, 0,
       B_*NP_, 128, 896, 1, 1, 1, NP_);
  cvtbf(stream, dglob, dglobh, (long)B_*128);
  gemm(stream, dglobh, 128, 0, wdgg0h, 128, 0, nullptr, dg1h, 512, 0, B_, 512, 128, 1, 1, 0, 0);
  gemm(stream, dg1h, 512, 0, wdgg1h, 512, 0, dg2, nullptr, 256, 0, B_, 256, 512, 1, 1, 0, 0);
  k_cls<<<(B_*40 + 255)/256, 256, 0, stream>>>(dg2, cls_w, cls_b, (float*)d_out);
}
